// DuaiMambaconf_34806414967223
// MI455X (gfx1250) — matrix operand order repaired and checked by reading, not compiled, not run on hardware
//
#include <hip/hip_runtime.h>
#include <hip/hip_bf16.h>
#include <math.h>

// ---------------------------------------------------------------------------
// CDNA5 (gfx1250) implementation of the dual-stream OSSM network.
// GEMMs use v_wmma_f32_16x16x32_bf16 (bf16 A/B staged through LDS, f32 acc).
// Fragments are stored contiguously per-lane in LDS -> ds_load_b128 path.
// ---------------------------------------------------------------------------

typedef __attribute__((ext_vector_type(16))) __bf16 v16bf;
typedef __attribute__((ext_vector_type(8)))  __bf16 v8bf;
typedef __attribute__((ext_vector_type(4)))  __bf16 v4bf;
typedef __attribute__((ext_vector_type(8)))  float  v8f;

union Frag16 { v16bf v; v8bf h[2]; };

#define ACT_NONE     0
#define ACT_GELU     1
#define ACT_SOFTPLUS 2

__device__ __forceinline__ float gelu_f(float x) {
    float x3 = x * x * x;
    return 0.5f * x * (1.0f + tanhf(0.7978845608028654f * (x + 0.044715f * x3)));
}
__device__ __forceinline__ float softplus_f(float x) {
    return (x > 20.0f) ? x : log1pf(__expf(x));
}
__device__ __forceinline__ float silu_f(float x) {
    return x / (1.0f + __expf(-x));
}

// ---------------------------------------------------------------------------
// NCHW (1,C,H,W) -> NHWC (H,W,C)
// ---------------------------------------------------------------------------
__global__ void k_nchw_nhwc(const float* __restrict__ src, float* __restrict__ dst,
                            int C, int H, int W) {
    int idx = blockIdx.x * blockDim.x + threadIdx.x;
    int total = C * H * W;
    if (idx >= total) return;
    int c = idx % C;
    int t = idx / C;
    dst[idx] = src[(size_t)c * H * W + t];
}

// ---------------------------------------------------------------------------
// Dense 3x3 conv, NHWC in, HWIO weights, pad=1
// ---------------------------------------------------------------------------
__global__ void k_conv3x3(const float* __restrict__ in, const float* __restrict__ w,
                          const float* __restrict__ bias, float* __restrict__ out,
                          int H, int W, int Cin, int Cout, int OH, int OW, int stride) {
    int idx = blockIdx.x * blockDim.x + threadIdx.x;
    int total = OH * OW * Cout;
    if (idx >= total) return;
    int co = idx % Cout;
    int t  = idx / Cout;
    int ox = t % OW, oy = t / OW;
    float acc = bias[co];
    #pragma unroll
    for (int ky = 0; ky < 3; ky++) {
        int iy = oy * stride + ky - 1;
        if (iy < 0 || iy >= H) continue;
        #pragma unroll
        for (int kx = 0; kx < 3; kx++) {
            int ix = ox * stride + kx - 1;
            if (ix < 0 || ix >= W) continue;
            const float* ip = in + ((size_t)iy * W + ix) * Cin;
            const float* wp = w + (size_t)((ky * 3 + kx) * Cin) * Cout + co;
            for (int ci = 0; ci < Cin; ci++)
                acc = fmaf(ip[ci], wp[(size_t)ci * Cout], acc);
        }
    }
    out[idx] = acc;
}

// ---------------------------------------------------------------------------
// Depthwise 3x3 conv (groups=C), stride 1, pad 1, fused SiLU
// ---------------------------------------------------------------------------
__global__ void k_dwconv_silu(const float* __restrict__ in, const float* __restrict__ w,
                              const float* __restrict__ bias, float* __restrict__ out,
                              int H, int W, int C) {
    int idx = blockIdx.x * blockDim.x + threadIdx.x;
    int total = H * W * C;
    if (idx >= total) return;
    int c = idx % C;
    int t = idx / C;
    int x = t % W, y = t / W;
    float acc = bias[c];
    #pragma unroll
    for (int ky = 0; ky < 3; ky++) {
        int iy = y + ky - 1;
        if (iy < 0 || iy >= H) continue;
        #pragma unroll
        for (int kx = 0; kx < 3; kx++) {
            int ix = x + kx - 1;
            if (ix < 0 || ix >= W) continue;
            acc = fmaf(in[((size_t)iy * W + ix) * C + c], w[(ky * 3 + kx) * C + c], acc);
        }
    }
    out[idx] = silu_f(acc);
}

// ---------------------------------------------------------------------------
// LayerNorm over last dim (one 128-thread block per row), optional fused GELU
// ---------------------------------------------------------------------------
__global__ __launch_bounds__(128) void k_ln(const float* __restrict__ x,
                                            const float* __restrict__ g,
                                            const float* __restrict__ b,
                                            float* __restrict__ out, int C, int fuse_gelu) {
    int row = blockIdx.x;
    const float* xr = x + (size_t)row * C;
    float* orow = out + (size_t)row * C;
    __shared__ float rs[128], rs2[128];
    float s = 0.f, s2 = 0.f;
    for (int c = threadIdx.x; c < C; c += 128) {
        float v = xr[c];
        s += v; s2 += v * v;
    }
    rs[threadIdx.x] = s; rs2[threadIdx.x] = s2;
    __syncthreads();
    for (int o = 64; o > 0; o >>= 1) {
        if (threadIdx.x < o) { rs[threadIdx.x] += rs[threadIdx.x + o]; rs2[threadIdx.x] += rs2[threadIdx.x + o]; }
        __syncthreads();
    }
    float mean = rs[0] / (float)C;
    float var  = rs2[0] / (float)C - mean * mean;
    float inv  = rsqrtf(var + 1e-5f);
    for (int c = threadIdx.x; c < C; c += 128) {
        float v = (xr[c] - mean) * inv * g[c] + b[c];
        orow[c] = fuse_gelu ? gelu_f(v) : v;
    }
}

// ---------------------------------------------------------------------------
// WMMA GEMM: C[M,N] = act(A[M,K] @ B[K,N] + bias)
// 64x64 macro-tile per 128-thread block (4 waves; wave w -> rows w*16..w*16+15,
// all 64 cols). A staged row-major, B staged TRANSPOSED so each lane's
// 16-element fragment is two contiguous 16B runs -> ds_load_b128.
// ---------------------------------------------------------------------------
#define TM 64
#define TN 64
#define TK 32
__global__ __launch_bounds__(128) void k_gemm(const float* __restrict__ A,
                                              const float* __restrict__ B,
                                              const float* __restrict__ bias,
                                              float* __restrict__ Cmat,
                                              int M, int N, int K, int act) {
    __shared__ __align__(16) __bf16 As[TM][TK];      // [m][k]
    __shared__ __align__(16) __bf16 Bst[TN][TK];     // transposed: [n][k]
    int tid  = threadIdx.x;
    int lane = tid & 31;
    int wave = tid >> 5;
    int m0 = blockIdx.y * TM;
    int n0 = blockIdx.x * TN;

    v8f acc[4];
    #pragma unroll
    for (int j = 0; j < 4; j++) {
        #pragma unroll
        for (int e = 0; e < 8; e++) acc[j][e] = 0.f;
    }

    int mrow  = lane & 15;   // A fragment row / B fragment column (within 16)
    int khalf = lane >> 4;   // which K half this lane holds
    int nkt = (K + TK - 1) / TK;

    for (int kt = 0; kt < nkt; kt++) {
        int k0 = kt * TK;

        // --- stage A tile (64x32): float4 global loads, packed v4bf LDS stores
        for (int i = tid * 4; i < TM * TK; i += 128 * 4) {
            int r = i >> 5, c = i & 31;          // c multiple of 4
            int gr = m0 + r, gc = k0 + c;
            float4 v = make_float4(0.f, 0.f, 0.f, 0.f);
            if (gr < M) {
                if (gc + 3 < K) {
                    v = *(const float4*)&A[(size_t)gr * K + gc];
                } else {
                    float t0 = (gc + 0 < K) ? A[(size_t)gr * K + gc + 0] : 0.f;
                    float t1 = (gc + 1 < K) ? A[(size_t)gr * K + gc + 1] : 0.f;
                    float t2 = (gc + 2 < K) ? A[(size_t)gr * K + gc + 2] : 0.f;
                    float t3 = (gc + 3 < K) ? A[(size_t)gr * K + gc + 3] : 0.f;
                    v = make_float4(t0, t1, t2, t3);
                }
            }
            v4bf p = { (__bf16)v.x, (__bf16)v.y, (__bf16)v.z, (__bf16)v.w };
            *(v4bf*)&As[r][c] = p;
        }
        // --- stage B tile (32x64) transposed: float4 loads along N, scalar
        //     transposed LDS stores
        for (int i = tid * 4; i < TK * TN; i += 128 * 4) {
            int r = i >> 6, c = i & 63;          // r = k, c = n (mult of 4)
            int gr = k0 + r, gc = n0 + c;
            float4 v = make_float4(0.f, 0.f, 0.f, 0.f);
            if (gr < K) {
                if (gc + 3 < N) {
                    v = *(const float4*)&B[(size_t)gr * N + gc];
                } else {
                    float t0 = (gc + 0 < N) ? B[(size_t)gr * N + gc + 0] : 0.f;
                    float t1 = (gc + 1 < N) ? B[(size_t)gr * N + gc + 1] : 0.f;
                    float t2 = (gc + 2 < N) ? B[(size_t)gr * N + gc + 2] : 0.f;
                    float t3 = (gc + 3 < N) ? B[(size_t)gr * N + gc + 3] : 0.f;
                    v = make_float4(t0, t1, t2, t3);
                }
            }
            Bst[c + 0][r] = (__bf16)v.x;
            Bst[c + 1][r] = (__bf16)v.y;
            Bst[c + 2][r] = (__bf16)v.z;
            Bst[c + 3][r] = (__bf16)v.w;
        }
        // --- prefetch next K tile (global_prefetch_b8) ---
        if (kt + 1 < nkt) {
            int k1 = k0 + TK;
            if (tid < 64) {
                int gr = m0 + tid;
                if (gr < M && k1 < K) __builtin_prefetch(&A[(size_t)gr * K + k1], 0, 1);
            } else {
                int gr = k1 + (tid - 64);
                if ((tid - 64) < TK && gr < K) __builtin_prefetch(&B[(size_t)gr * N + n0], 0, 1);
            }
        }
        __syncthreads();

        // --- A fragment: two contiguous 16B runs -> 2x ds_load_b128
        Frag16 af;
        int mw = wave * 16 + mrow;
        af.h[0] = *(const v8bf*)&As[mw][khalf * 8];
        af.h[1] = *(const v8bf*)&As[mw][16 + khalf * 8];

        // --- 4 x (B fragment + WMMA); B fragment = 2x ds_load_b128
        #pragma unroll
        for (int j = 0; j < 4; j++) {
            Frag16 bf_;
            const __bf16* bp = &Bst[j * 16 + mrow][khalf * 8];
            bf_.h[0] = *(const v8bf*)bp;
            bf_.h[1] = *(const v8bf*)(bp + 16);
            acc[j] = __builtin_amdgcn_wmma_f32_16x16x32_bf16(
                false, af.v, false, bf_.v, (short)0, acc[j], false, false);
        }
        __syncthreads();
    }

    // --- epilogue: C/D layout: VGPR r -> M = r (lanes 0-15) / 8+r (16-31)
    int nl = lane & 15;
    int mb = (lane < 16) ? 0 : 8;
    #pragma unroll
    for (int j = 0; j < 4; j++) {
        #pragma unroll
        for (int r = 0; r < 8; r++) {
            int row = m0 + wave * 16 + mb + r;
            int col = n0 + j * 16 + nl;
            if (row < M && col < N) {
                float v = acc[j][r];
                if (bias) v += bias[col];
                if (act == ACT_GELU)          v = gelu_f(v);
                else if (act == ACT_SOFTPLUS) v = softplus_f(v);
                Cmat[(size_t)row * N + col] = v;
            }
        }
    }
}

// ---------------------------------------------------------------------------
// Selective scan: one wave per 2 channels; lane = (c_sub<<4) | n, N=16 states.
// ---------------------------------------------------------------------------
__global__ __launch_bounds__(32) void k_scan(const float* __restrict__ u,
                                             const float* __restrict__ dt,
                                             const float* __restrict__ A_log,
                                             const float* __restrict__ Bm,
                                             const float* __restrict__ Cm,
                                             const float* __restrict__ D,
                                             float* __restrict__ y, int T, int Ci) {
    int lane = threadIdx.x;
    int n = lane & 15;
    int c = blockIdx.x * 2 + (lane >> 4);
    float A  = -__expf(A_log[(size_t)c * 16 + n]);
    float Dc = D[c];
    float h = 0.f;
    for (int t = 0; t < T; t++) {
        float dtv = dt[(size_t)t * Ci + c];
        float uv  = u[(size_t)t * Ci + c];
        float bv  = Bm[(size_t)t * 16 + n];
        float cv  = Cm[(size_t)t * 16 + n];
        h = __expf(dtv * A) * h + dtv * uv * bv;
        float s = h * cv;
        s += __shfl_xor(s, 8, 16);
        s += __shfl_xor(s, 4, 16);
        s += __shfl_xor(s, 2, 16);
        s += __shfl_xor(s, 1, 16);
        if (n == 0) y[(size_t)t * Ci + c] = s + uv * Dc;
    }
}

// ---------------------------------------------------------------------------
// Elementwise helpers
// ---------------------------------------------------------------------------
__global__ void k_add(const float* __restrict__ a, const float* __restrict__ b,
                      float* __restrict__ out, int n) {
    int i = blockIdx.x * blockDim.x + threadIdx.x;
    if (i < n) out[i] = a[i] + b[i];
}
__global__ void k_mul_silu(const float* __restrict__ a, const float* __restrict__ z,
                           float* __restrict__ out, int n) {
    int i = blockIdx.x * blockDim.x + threadIdx.x;
    if (i < n) out[i] = a[i] * silu_f(z[i]);
}
__global__ void k_copy_cols(const float* __restrict__ src, float* __restrict__ dst,
                            int M, int Cs, int Cd, int scol0, int dcol0, int Cn) {
    int i = blockIdx.x * blockDim.x + threadIdx.x;
    if (i >= M * Cn) return;
    int c = i % Cn, r = i / Cn;
    dst[(size_t)r * Cd + dcol0 + c] = src[(size_t)r * Cs + scol0 + c];
}

// ---------------------------------------------------------------------------
// Host orchestration
// ---------------------------------------------------------------------------
struct OssmP {
    const float *in_w, *in_b, *conv_w, *conv_b, *xproj_w, *dt_w, *dt_b;
    const float *A_log, *D, *onorm_g, *onorm_b, *out_w, *out_b;
};
struct BlockP {
    const float *n1g, *n1b, *n2g, *n2b;
    OssmP ossm;
    const float *w1, *b1, *w2, *b2;
};

extern "C" void kernel_launch(void* const* d_in, const int* in_sizes, int n_in,
                              void* d_out, int out_size, void* d_ws, size_t ws_size,
                              hipStream_t stream) {
    (void)in_sizes; (void)n_in; (void)out_size; (void)ws_size;

    int ip = 0;
    auto nxt = [&]() { return (const float*)d_in[ip++]; };
    const float* in_x = nxt();
    const float* in_y = nxt();
    const float *c1w = nxt(), *c1b = nxt(), *l1g = nxt(), *l1b = nxt();
    const float *c2w = nxt(), *c2b = nxt(), *l2g = nxt(), *l2b = nxt();
    struct { const float *cw, *cb, *lg, *lb; } ds[3];
    for (int i = 0; i < 3; i++) { ds[i].cw = nxt(); ds[i].cb = nxt(); ds[i].lg = nxt(); ds[i].lb = nxt(); }
    static const int dims[4]   = {64, 128, 256, 512};
    static const int depths[4] = {3, 4, 6, 3};
    BlockP blocks[4][6];
    for (int s = 0; s < 4; s++) {
        for (int bI = 0; bI < depths[s]; bI++) {
            BlockP& B = blocks[s][bI];
            B.n1g = nxt(); B.n1b = nxt(); B.n2g = nxt(); B.n2b = nxt();
            OssmP& o = B.ossm;
            o.in_w = nxt(); o.in_b = nxt(); o.conv_w = nxt(); o.conv_b = nxt();
            o.xproj_w = nxt(); o.dt_w = nxt(); o.dt_b = nxt();
            o.A_log = nxt(); o.D = nxt(); o.onorm_g = nxt(); o.onorm_b = nxt();
            o.out_w = nxt(); o.out_b = nxt();
            B.w1 = nxt(); B.b1 = nxt(); B.w2 = nxt(); B.b2 = nxt();
        }
    }
    const float* norm_g[4]; const float* norm_b[4];
    for (int s = 0; s < 4; s++) { norm_g[s] = nxt(); norm_b[s] = nxt(); }

    float* wf = (float*)d_ws;
    size_t off = 0;
    auto alloc = [&](size_t nflt) { float* p = wf + off; off += nflt; return p; };
    float* BX   = alloc(262144);
    float* BY   = alloc(262144);
    float* T1   = alloc(262144);
    float* XZ   = alloc(1048576);
    float* XS   = alloc(524288);
    float* ZB   = alloc(524288);
    float* UB   = alloc(524288);
    float* DTB  = alloc(524288);
    float* DBL  = alloc(294912);
    float* DTIN = alloc(131072);
    float* BMB  = alloc(65536);
    float* CMB  = alloc(65536);
    float* YSB  = alloc(524288);
    float* HB   = alloc(131072);
    float* OB   = alloc(262144);
    float* NH   = alloc(196608);
    float* PE   = alloc(524288);

    auto cdiv = [](int a, int b) { return (a + b - 1) / b; };
    auto gemm = [&](const float* A, const float* B, const float* bias, float* C,
                    int M, int N, int K, int act) {
        dim3 grid(cdiv(N, TN), cdiv(M, TM));
        k_gemm<<<grid, 128, 0, stream>>>(A, B, bias, C, M, N, K, act);
    };
    auto ln = [&](const float* x, const float* g, const float* b, float* out,
                  int T, int C, int fuse) {
        k_ln<<<T, 128, 0, stream>>>(x, g, b, out, C, fuse);
    };
    auto ew = [&](int n) { return dim3(cdiv(n, 256)); };

    auto patch_embed = [&](const float* src, float* act) {
        k_nchw_nhwc<<<ew(3 * 256 * 256), 256, 0, stream>>>(src, NH, 3, 256, 256);
        k_conv3x3<<<ew(128 * 128 * 32), 256, 0, stream>>>(NH, c1w, c1b, PE, 256, 256, 3, 32, 128, 128, 2);
        ln(PE, l1g, l1b, PE, 128 * 128, 32, 1);
        k_conv3x3<<<ew(64 * 64 * 64), 256, 0, stream>>>(PE, c2w, c2b, act, 128, 128, 32, 64, 64, 64, 2);
        ln(act, l2g, l2b, act, 4096, 64, 0);
    };
    patch_embed(in_x, BX);
    patch_embed(in_y, BY);

    auto run_ossm = [&](const float* xin, float* outp, const OssmP& p,
                        int T, int C, int H, int W) {
        int Ci = 2 * C, R = C / 16, Nx = R + 32;
        gemm(xin, p.in_w, p.in_b, XZ, T, 4 * C, C, ACT_NONE);
        k_copy_cols<<<ew(T * Ci), 256, 0, stream>>>(XZ, XS, T, 4 * C, Ci, 0, 0, Ci);
        k_copy_cols<<<ew(T * Ci), 256, 0, stream>>>(XZ, ZB, T, 4 * C, Ci, Ci, 0, Ci);
        k_dwconv_silu<<<ew(T * Ci), 256, 0, stream>>>(XS, p.conv_w, p.conv_b, UB, H, W, Ci);
        gemm(UB, p.xproj_w, nullptr, DBL, T, Nx, Ci, ACT_NONE);
        k_copy_cols<<<ew(T * R), 256, 0, stream>>>(DBL, DTIN, T, Nx, R, 0, 0, R);
        k_copy_cols<<<ew(T * 16), 256, 0, stream>>>(DBL, BMB, T, Nx, 16, R, 0, 16);
        k_copy_cols<<<ew(T * 16), 256, 0, stream>>>(DBL, CMB, T, Nx, 16, R + 16, 0, 16);
        gemm(DTIN, p.dt_w, p.dt_b, DTB, T, Ci, R, ACT_SOFTPLUS);
        k_scan<<<Ci / 2, 32, 0, stream>>>(UB, DTB, p.A_log, BMB, CMB, p.D, YSB, T, Ci);
        ln(YSB, p.onorm_g, p.onorm_b, YSB, T, Ci, 0);
        k_mul_silu<<<ew(T * Ci), 256, 0, stream>>>(YSB, ZB, YSB, T * Ci);
        gemm(YSB, p.out_w, p.out_b, outp, T, C, Ci, ACT_NONE);
    };
    auto run_block_stream = [&](float* act, const BlockP& B, int T, int C, int H, int W) {
        ln(act, B.n1g, B.n1b, T1, T, C, 0);
        run_ossm(T1, OB, B.ossm, T, C, H, W);
        k_add<<<ew(T * C), 256, 0, stream>>>(act, OB, act, T * C);
        ln(act, B.n2g, B.n2b, T1, T, C, 0);
        gemm(T1, B.w1, B.b1, HB, T, C / 2, C, ACT_GELU);
        gemm(HB, B.w2, B.b2, OB, T, C, C / 2, ACT_NONE);
        k_add<<<ew(T * C), 256, 0, stream>>>(act, OB, act, T * C);
    };

    for (int s = 0; s < 4; s++) {
        int H = 64 >> s, W = H, C = dims[s], T = H * W;
        if (s > 0) {
            int Hp = 64 >> (s - 1), Wp = Hp, Cp = dims[s - 1];
            k_conv3x3<<<ew(T * C), 256, 0, stream>>>(BX, ds[s-1].cw, ds[s-1].cb, XZ, Hp, Wp, Cp, C, H, W, 2);
            ln(XZ, ds[s-1].lg, ds[s-1].lb, XZ, T, C, 0);
            hipMemcpyAsync(BX, XZ, (size_t)T * C * sizeof(float), hipMemcpyDeviceToDevice, stream);
            k_conv3x3<<<ew(T * C), 256, 0, stream>>>(BY, ds[s-1].cw, ds[s-1].cb, XZ, Hp, Wp, Cp, C, H, W, 2);
            ln(XZ, ds[s-1].lg, ds[s-1].lb, XZ, T, C, 0);
            hipMemcpyAsync(BY, XZ, (size_t)T * C * sizeof(float), hipMemcpyDeviceToDevice, stream);
        }
        for (int bI = 0; bI < depths[s]; bI++) {
            run_block_stream(BX, blocks[s][bI], T, C, H, W);
            run_block_stream(BY, blocks[s][bI], T, C, H, W);
        }
        ln(BX, norm_g[s], norm_b[s], BX, T, C, 0);
        ln(BY, norm_g[s], norm_b[s], BY, T, C, 0);
    }

    float* out = (float*)d_out;
    k_copy_cols<<<ew(64 * 512), 256, 0, stream>>>(BX, out, 64, 512, 1024, 0, 0, 512);
    k_copy_cols<<<ew(64 * 512), 256, 0, stream>>>(BY, out, 64, 512, 1024, 0, 512, 512);
}
